// GraphClassifier_54150947668734
// MI455X (gfx1250) — matrix operand order repaired and checked by reading, not compiled, not run on hardware
//
#include <hip/hip_runtime.h>
#include <math.h>

#define DEV __device__ __forceinline__

typedef __attribute__((ext_vector_type(16))) __bf16 v16bf;
typedef __attribute__((ext_vector_type(8)))  float  v8f;

union ABFrag { v16bf v; unsigned u[8]; };

static constexpr int B_   = 32;
static constexpr int NPER = 1024;
static constexpr int EPER = 16384;
static constexpr int N_   = B_ * NPER;   // 32768
static constexpr int E_   = B_ * EPER;   // 524288

DEV unsigned short f2bf(float f) {
  unsigned u = __float_as_uint(f);
  unsigned r = u + 0x7FFFu + ((u >> 16) & 1u);
  return (unsigned short)(r >> 16);
}
DEV unsigned packbf(float a, float b) {
  return (unsigned)f2bf(a) | ((unsigned)f2bf(b) << 16);
}
DEV float sigmoidf_(float x) { return 1.0f / (1.0f + expf(-x)); }

DEV v8f wmma_bf16(const ABFrag& a, const ABFrag& b, v8f c) {
  return __builtin_amdgcn_wmma_f32_16x16x32_bf16(false, a.v, false, b.v,
                                                 (short)0, c, false, false);
}

// A fragment: 16x32 bf16 tile. lane&15 = M row; lane>>4 selects K halves:
// VGPR0..7 hold K = g*16 + {0..15} (contiguous, matching load_b).
DEV void load_a(ABFrag& f, const unsigned* ldsA, int rowBase, int strideU,
                int ktile, int lane) {
  const int m = lane & 15, g = lane >> 4;
  const unsigned* rp = ldsA + (size_t)(rowBase + m) * strideU + ktile * 16 + g * 8;
#pragma unroll
  for (int p = 0; p < 4; ++p) f.u[p] = rp[p];
#pragma unroll
  for (int p = 0; p < 4; ++p) f.u[4 + p] = rp[4 + p];
}
// B fragment: 32x16 bf16 (KxN). lane&15 = N column; lanes 0-15 K=0..15, 16-31 K=16..31.
// ldsW is W row-major [64][K] bf16 pairs; B[k][n] = W[n][k].
DEV void load_b(ABFrag& f, const unsigned* ldsW, int strideU, int ntile,
                int ktile, int lane) {
  const int n = lane & 15, g = lane >> 4;
  const unsigned* rp = ldsW + (size_t)(ntile * 16 + n) * strideU + ktile * 16 + g * 8;
#pragma unroll
  for (int p = 0; p < 8; ++p) f.u[p] = rp[p];
}

// ---------------- input_node = relu(node_feat @ W_i_node^T); copy to message_node
__global__ void __launch_bounds__(256) k_input_node(const float* __restrict__ X,
                                                    const float* __restrict__ W,
                                                    float* __restrict__ out1,
                                                    float* __restrict__ out2) {
  __shared__ unsigned sA[128 * 32];
  __shared__ unsigned sW[64 * 32];
  const int tid = threadIdx.x, lane = tid & 31, wave = tid >> 5;
  const size_t base = (size_t)blockIdx.x * 128 * 64;
  for (int i = tid; i < 64 * 32; i += 256) sW[i] = packbf(W[2 * i], W[2 * i + 1]);
  for (int i = tid; i < 128 * 32; i += 256)
    sA[i] = packbf(X[base + 2 * i], X[base + 2 * i + 1]);
  __syncthreads();
  ABFrag a0, a1;
  load_a(a0, sA, wave * 16, 32, 0, lane);
  load_a(a1, sA, wave * 16, 32, 1, lane);
  const int mb = (lane >> 4) * 8, n = lane & 15;
#pragma unroll
  for (int nt = 0; nt < 4; ++nt) {
    v8f c = {0.f, 0.f, 0.f, 0.f, 0.f, 0.f, 0.f, 0.f};
    ABFrag bf;
    load_b(bf, sW, 32, nt, 0, lane); c = wmma_bf16(a0, bf, c);
    load_b(bf, sW, 32, nt, 1, lane); c = wmma_bf16(a1, bf, c);
#pragma unroll
    for (int r = 0; r < 8; ++r) {
      float v = fmaxf(c[r], 0.f);
      size_t o = base + (size_t)(wave * 16 + mb + r) * 64 + nt * 16 + n;
      out1[o] = v; out2[o] = v;
    }
  }
}

// ---------------- per-graph: target_rel, tgt_rel_h
__global__ void __launch_bounds__(64) k_graph_prep(const int* __restrict__ glabel,
                                                   const float* __restrict__ rel_emb,
                                                   const float* __restrict__ W_r2e,
                                                   const float* __restrict__ b_r2e,
                                                   float* __restrict__ target_rel,
                                                   float* __restrict__ tgt_rel_h) {
  const int b = blockIdx.x, j = threadIdx.x;
  __shared__ float tr[64];
  float v = rel_emb[(size_t)glabel[b] * 64 + j];
  tr[j] = v;
  target_rel[b * 64 + j] = v;
  __syncthreads();
  float s = b_r2e[j];
  for (int k = 0; k < 64; ++k) s += tr[k] * W_r2e[j * 64 + k];
  tgt_rel_h[b * 64 + j] = s;
}

// ---------------- per-graph: gpart = ge @ W_att1[:,64:128]^T
__global__ void __launch_bounds__(64) k_gpart(const int* __restrict__ srcn,
                                              const int* __restrict__ tgtn,
                                              const float* __restrict__ mn,
                                              const float* __restrict__ trh,
                                              const float* __restrict__ W_att1,
                                              float* __restrict__ gpart) {
  const int b = blockIdx.x, j = threadIdx.x;
  __shared__ float ge[64];
  const int s = srcn[b], t = tgtn[b];
  ge[j] = mn[(size_t)s * 64 + j] + trh[b * 64 + j] - mn[(size_t)t * 64 + j];
  __syncthreads();
  float acc = 0.f;
  for (int k = 0; k < 64; ++k) acc += ge[k] * W_att1[j * 128 + 64 + k];
  gpart[b * 64 + j] = acc;
}

// ---------------- fused: input_edge = relu(edge_feat @ W_i_edge^T); att0 = sigmoid(relu(ie@W1a^T+gpart)@w2)
__global__ void __launch_bounds__(256) k_edge_init(
    const float* __restrict__ node_feat, const float* __restrict__ rel_emb,
    const int* __restrict__ esrc, const int* __restrict__ edst,
    const int* __restrict__ etype, const float* __restrict__ W_ie,
    const float* __restrict__ W_att1, const float* __restrict__ W_att2,
    const float* __restrict__ gpart, float* __restrict__ input_edge,
    float* __restrict__ att0) {
  __shared__ unsigned sA[128 * 96];   // 48KB; reused (front) for hidden stage
  __shared__ unsigned sW[64 * 96];    // W_i_edge bf16
  __shared__ unsigned sW1a[64 * 32];
  __shared__ float sw2[64];
  __shared__ float sgp[64];
  __shared__ int sSrc[128], sDst[128], sEt[128];
  const int tid = threadIdx.x, lane = tid & 31, wave = tid >> 5;
  const int ebase = blockIdx.x * 128;
  const int g = ebase / EPER;
  for (int i = tid; i < 64 * 96; i += 256) sW[i] = packbf(W_ie[2 * i], W_ie[2 * i + 1]);
  for (int i = tid; i < 64 * 32; i += 256) {
    int r = i >> 5, p = i & 31;
    sW1a[i] = packbf(W_att1[r * 128 + 2 * p], W_att1[r * 128 + 2 * p + 1]);
  }
  if (tid < 64) { sw2[tid] = W_att2[tid]; sgp[tid] = gpart[g * 64 + tid]; }
  if (tid < 128) {
    int e = ebase + tid;
    sSrc[tid] = esrc[e]; sDst[tid] = edst[e]; sEt[tid] = etype[e];
  }
  __syncthreads();
  for (int i = tid; i < 128 * 96; i += 256) {
    int row = i / 96, p = i % 96, k = 2 * p;
    float a, b2;
    if (k < 64)       { const float* r0 = node_feat + (size_t)sSrc[row] * 64 + k;        a = r0[0]; b2 = r0[1]; }
    else if (k < 128) { const float* r0 = rel_emb   + (size_t)sEt[row]  * 64 + (k - 64); a = r0[0]; b2 = r0[1]; }
    else              { const float* r0 = node_feat + (size_t)sDst[row] * 64 + (k - 128);a = r0[0]; b2 = r0[1]; }
    sA[i] = packbf(a, b2);
  }
  __syncthreads();
  ABFrag afr[6];
#pragma unroll
  for (int kt = 0; kt < 6; ++kt) load_a(afr[kt], sA, wave * 16, 96, kt, lane);
  const int mb = (lane >> 4) * 8, n = lane & 15;
  float ie[4][8];
#pragma unroll
  for (int nt = 0; nt < 4; ++nt) {
    v8f c = {0.f, 0.f, 0.f, 0.f, 0.f, 0.f, 0.f, 0.f};
#pragma unroll
    for (int kt = 0; kt < 6; ++kt) {
      ABFrag bf; load_b(bf, sW, 96, nt, kt, lane);
      c = wmma_bf16(afr[kt], bf, c);
    }
#pragma unroll
    for (int r = 0; r < 8; ++r) {
      float v = fmaxf(c[r], 0.f);
      ie[nt][r] = v;
      input_edge[(size_t)(ebase + wave * 16 + mb + r) * 64 + nt * 16 + n] = v;
    }
  }
  __syncthreads();
  unsigned short* sAh = (unsigned short*)sA;   // hidden A panel: 128 x 64 bf16
#pragma unroll
  for (int nt = 0; nt < 4; ++nt)
#pragma unroll
    for (int r = 0; r < 8; ++r)
      sAh[(size_t)(wave * 16 + mb + r) * 64 + nt * 16 + n] = f2bf(ie[nt][r]);
  __syncthreads();
  ABFrag h0, h1;
  load_a(h0, sA, wave * 16, 32, 0, lane);
  load_a(h1, sA, wave * 16, 32, 1, lane);
  float part[8];
#pragma unroll
  for (int r = 0; r < 8; ++r) part[r] = 0.f;
#pragma unroll
  for (int nt = 0; nt < 4; ++nt) {
    float gv = sgp[nt * 16 + n];
    v8f c = {gv, gv, gv, gv, gv, gv, gv, gv};
    ABFrag bf;
    load_b(bf, sW1a, 32, nt, 0, lane); c = wmma_bf16(h0, bf, c);
    load_b(bf, sW1a, 32, nt, 1, lane); c = wmma_bf16(h1, bf, c);
    float w2v = sw2[nt * 16 + n];
#pragma unroll
    for (int r = 0; r < 8; ++r) part[r] += fmaxf(c[r], 0.f) * w2v;
  }
#pragma unroll
  for (int r = 0; r < 8; ++r) {
    float p = part[r];
    p += __shfl_xor(p, 1); p += __shfl_xor(p, 2);
    p += __shfl_xor(p, 4); p += __shfl_xor(p, 8);
    part[r] = p;
  }
  if ((lane & 15) < 8) {
    int r = lane & 7;
    att0[ebase + wave * 16 + mb + r] = sigmoidf_(part[r]);
  }
}

// ---------------- scatter-add: agg[dst] += message_edge * att
__global__ void __launch_bounds__(256) k_scatter(const float* __restrict__ me,
                                                 const float* __restrict__ att_e,
                                                 const float* __restrict__ att_g,
                                                 const int* __restrict__ edst,
                                                 float* __restrict__ agg) {
  const unsigned idx = blockIdx.x * 256u + threadIdx.x;
  const int e = idx >> 6, k = idx & 63;
  float a = att_e ? att_e[e] : att_g[e >> 14];   // EPER == 2^14
  atomicAdd(&agg[(size_t)edst[e] * 64 + k], me[idx] * a);
}

// ---------------- message_node = relu((mn + agg) @ W^T)
__global__ void __launch_bounds__(256) k_node_update(const float* __restrict__ mn,
                                                     const float* __restrict__ agg,
                                                     const float* __restrict__ W,
                                                     float* __restrict__ out) {
  __shared__ unsigned sA[128 * 32];
  __shared__ unsigned sW[64 * 32];
  const int tid = threadIdx.x, lane = tid & 31, wave = tid >> 5;
  const size_t base = (size_t)blockIdx.x * 128 * 64;
  for (int i = tid; i < 64 * 32; i += 256) sW[i] = packbf(W[2 * i], W[2 * i + 1]);
  for (int i = tid; i < 128 * 32; i += 256)
    sA[i] = packbf(mn[base + 2 * i] + agg[base + 2 * i],
                   mn[base + 2 * i + 1] + agg[base + 2 * i + 1]);
  __syncthreads();
  ABFrag a0, a1;
  load_a(a0, sA, wave * 16, 32, 0, lane);
  load_a(a1, sA, wave * 16, 32, 1, lane);
  const int mb = (lane >> 4) * 8, n = lane & 15;
#pragma unroll
  for (int nt = 0; nt < 4; ++nt) {
    v8f c = {0.f, 0.f, 0.f, 0.f, 0.f, 0.f, 0.f, 0.f};
    ABFrag bf;
    load_b(bf, sW, 32, nt, 0, lane); c = wmma_bf16(a0, bf, c);
    load_b(bf, sW, 32, nt, 1, lane); c = wmma_bf16(a1, bf, c);
#pragma unroll
    for (int r = 0; r < 8; ++r)
      out[base + (size_t)(wave * 16 + mb + r) * 64 + nt * 16 + n] = fmaxf(c[r], 0.f);
  }
}

// ---------------- message_edge = relu(input_edge + mn[src] @ W^T)
__global__ void __launch_bounds__(256) k_edge_update(const int* __restrict__ esrc,
                                                     const float* __restrict__ mn,
                                                     const float* __restrict__ ie,
                                                     const float* __restrict__ W,
                                                     float* __restrict__ me) {
  __shared__ unsigned sA[128 * 32];
  __shared__ unsigned sW[64 * 32];
  __shared__ int sSrc[128];
  const int tid = threadIdx.x, lane = tid & 31, wave = tid >> 5;
  const int ebase = blockIdx.x * 128;
  for (int i = tid; i < 64 * 32; i += 256) sW[i] = packbf(W[2 * i], W[2 * i + 1]);
  if (tid < 128) sSrc[tid] = esrc[ebase + tid];
  __syncthreads();
  for (int i = tid; i < 128 * 32; i += 256) {
    int row = i >> 5, p = i & 31;
    const float* r0 = mn + (size_t)sSrc[row] * 64 + 2 * p;
    sA[i] = packbf(r0[0], r0[1]);
  }
  __syncthreads();
  ABFrag a0, a1;
  load_a(a0, sA, wave * 16, 32, 0, lane);
  load_a(a1, sA, wave * 16, 32, 1, lane);
  const int mb = (lane >> 4) * 8, n = lane & 15;
#pragma unroll
  for (int nt = 0; nt < 4; ++nt) {
    v8f c = {0.f, 0.f, 0.f, 0.f, 0.f, 0.f, 0.f, 0.f};
    ABFrag bf;
    load_b(bf, sW, 32, nt, 0, lane); c = wmma_bf16(a0, bf, c);
    load_b(bf, sW, 32, nt, 1, lane); c = wmma_bf16(a1, bf, c);
#pragma unroll
    for (int r = 0; r < 8; ++r) {
      size_t o = (size_t)(ebase + wave * 16 + mb + r) * 64 + nt * 16 + n;
      me[o] = fmaxf(c[r] + ie[o], 0.f);
    }
  }
}

// ---------------- per-graph attention for loop iterations
__global__ void __launch_bounds__(64) k_graph_att(const int* __restrict__ srcn,
                                                  const int* __restrict__ tgtn,
                                                  const float* __restrict__ mn,
                                                  const float* __restrict__ trh,
                                                  const float* __restrict__ trel,
                                                  const float* __restrict__ WA1,
                                                  const float* __restrict__ WA2,
                                                  float* __restrict__ attg) {
  const int b = blockIdx.x, j = threadIdx.x;
  __shared__ float ge[64];
  __shared__ float h1[64];
  const int s = srcn[b], t = tgtn[b];
  ge[j] = mn[(size_t)s * 64 + j] + trh[b * 64 + j] - mn[(size_t)t * 64 + j];
  __syncthreads();
  float acc = 0.f;
  for (int k = 0; k < 64; ++k) acc += ge[k] * WA1[j * 128 + k];
  for (int k = 0; k < 64; ++k) acc += trel[b * 64 + k] * WA1[j * 128 + 64 + k];
  h1[j] = fmaxf(acc, 0.f);
  __syncthreads();
  if (j == 0) {
    float s2 = 0.f;
    for (int k = 0; k < 64; ++k) s2 += h1[k] * WA2[k];
    attg[b] = sigmoidf_(s2);
  }
}

// ---------------- node_h = concat(agg, mn, input_node) @ W_comm^T
__global__ void __launch_bounds__(256) k_comm(const float* __restrict__ agg,
                                              const float* __restrict__ mn,
                                              const float* __restrict__ inode,
                                              const float* __restrict__ Wc,
                                              float* __restrict__ node_h) {
  __shared__ unsigned sA[128 * 96];
  __shared__ unsigned sW[64 * 96];
  const int tid = threadIdx.x, lane = tid & 31, wave = tid >> 5;
  const int rowbase = blockIdx.x * 128;
  for (int i = tid; i < 64 * 96; i += 256) sW[i] = packbf(Wc[2 * i], Wc[2 * i + 1]);
  for (int i = tid; i < 128 * 96; i += 256) {
    int row = i / 96, p = i % 96, k = 2 * p;
    size_t gr = (size_t)(rowbase + row) * 64;
    const float* src = (k < 64) ? (agg + gr + k)
                     : (k < 128) ? (mn + gr + (k - 64))
                                 : (inode + gr + (k - 128));
    sA[i] = packbf(src[0], src[1]);
  }
  __syncthreads();
  ABFrag afr[6];
#pragma unroll
  for (int kt = 0; kt < 6; ++kt) load_a(afr[kt], sA, wave * 16, 96, kt, lane);
  const int mb = (lane >> 4) * 8, n = lane & 15;
#pragma unroll
  for (int nt = 0; nt < 4; ++nt) {
    v8f c = {0.f, 0.f, 0.f, 0.f, 0.f, 0.f, 0.f, 0.f};
#pragma unroll
    for (int kt = 0; kt < 6; ++kt) {
      ABFrag bf; load_b(bf, sW, 96, nt, kt, lane);
      c = wmma_bf16(afr[kt], bf, c);
    }
#pragma unroll
    for (int r = 0; r < 8; ++r)
      node_h[(size_t)(rowbase + wave * 16 + mb + r) * 64 + nt * 16 + n] = c[r];
  }
}

// ---------------- h0 = per-graph max of node_h
__global__ void __launch_bounds__(256) k_h0(const float* __restrict__ node_h,
                                            float* __restrict__ h0) {
  const int b = blockIdx.x, tid = threadIdx.x;
  const int col = tid & 63, seg = tid >> 6;
  float m = -3.402823466e+38f;
  for (int r = seg; r < NPER; r += 4)
    m = fmaxf(m, node_h[(size_t)(b * NPER + r) * 64 + col]);
  __shared__ float red[256];
  red[tid] = m;
  __syncthreads();
  if (tid < 64) {
    float v = fmaxf(fmaxf(red[tid], red[tid + 64]), fmaxf(red[tid + 128], red[tid + 192]));
    h0[b * 64 + tid] = v;
  }
}

// ---------------- GI[dir] = relu(node_h + gru_bias) @ W_ih[dir]^T + b_ih[dir]
__global__ void __launch_bounds__(256) k_gi(const float* __restrict__ node_h,
                                            const float* __restrict__ gbias,
                                            const float* __restrict__ Wih,
                                            const float* __restrict__ bih,
                                            float* __restrict__ GI) {
  __shared__ unsigned sA[128 * 32];
  __shared__ unsigned sW[192 * 32];
  __shared__ float sb[192];
  __shared__ float sbias[64];
  const int tid = threadIdx.x, lane = tid & 31, wave = tid >> 5;
  const int rowbase = blockIdx.x * 128;
  const int dir = blockIdx.y;
  const float* Wd = Wih + (size_t)dir * 192 * 64;
  float* GId = GI + (size_t)dir * N_ * 192;
  for (int i = tid; i < 192 * 32; i += 256) sW[i] = packbf(Wd[2 * i], Wd[2 * i + 1]);
  if (tid < 192) sb[tid] = bih[dir * 192 + tid];
  if (tid < 64) sbias[tid] = gbias[tid];
  __syncthreads();
  for (int i = tid; i < 128 * 32; i += 256) {
    int row = i >> 5, k = 2 * (i & 31);
    const float* r0 = node_h + (size_t)(rowbase + row) * 64;
    sA[i] = packbf(fmaxf(r0[k] + sbias[k], 0.f), fmaxf(r0[k + 1] + sbias[k + 1], 0.f));
  }
  __syncthreads();
  ABFrag a0, a1;
  load_a(a0, sA, wave * 16, 32, 0, lane);
  load_a(a1, sA, wave * 16, 32, 1, lane);
  const int mb = (lane >> 4) * 8, n = lane & 15;
  for (int nt = 0; nt < 12; ++nt) {
    float bv = sb[nt * 16 + n];
    v8f c = {bv, bv, bv, bv, bv, bv, bv, bv};
    ABFrag bf;
    load_b(bf, sW, 32, nt, 0, lane); c = wmma_bf16(a0, bf, c);
    load_b(bf, sW, 32, nt, 1, lane); c = wmma_bf16(a1, bf, c);
#pragma unroll
    for (int r = 0; r < 8; ++r)
      GId[(size_t)(rowbase + wave * 16 + mb + r) * 192 + nt * 16 + n] = c[r];
  }
}

// ---------------- GRU recurrence: one block per (graph, direction)
__global__ void __launch_bounds__(64) k_gru_seq(const float* __restrict__ GI,
                                                const float* __restrict__ Whh,
                                                const float* __restrict__ bhh,
                                                const float* __restrict__ h0,
                                                float* __restrict__ gout) {
  const int b = blockIdx.x, dir = blockIdx.y, j = threadIdx.x;
  __shared__ float sW[192 * 64];
  __shared__ float sh[64];
  __shared__ float sb[192];
  const float* W = Whh + (size_t)dir * 192 * 64;
  for (int i = j; i < 192 * 64; i += 64) sW[i] = W[i];
  for (int i = j; i < 192; i += 64) sb[i] = bhh[dir * 192 + i];
  sh[j] = h0[b * 64 + j];
  __syncthreads();
  const float* gid = GI + (size_t)dir * N_ * 192;
  for (int t = 0; t < NPER; ++t) {
    const int nidx = b * NPER + (dir ? (NPER - 1 - t) : t);
    const float* gi = gid + (size_t)nidx * 192;
    float gr = gi[j], gz = gi[64 + j], gn = gi[128 + j];
    float hr = sb[j], hz = sb[64 + j], hn = sb[128 + j];
    for (int k = 0; k < 64; ++k) {
      float hk = sh[k];
      hr += hk * sW[j * 64 + k];
      hz += hk * sW[(64 + j) * 64 + k];
      hn += hk * sW[(128 + j) * 64 + k];
    }
    float r = sigmoidf_(gr + hr);
    float z = sigmoidf_(gz + hz);
    float nn = tanhf(gn + r * hn);
    float hnew = (1.f - z) * nn + z * sh[j];
    __syncthreads();
    sh[j] = hnew;
    __syncthreads();
    gout[(size_t)nidx * 128 + dir * 64 + j] = hnew;
  }
}

// ---------------- final head: atom_h at src/tgt, conv, two tiny linears
__global__ void __launch_bounds__(64) k_final(const int* __restrict__ srcn,
                                              const int* __restrict__ tgtn,
                                              const float* __restrict__ gout,
                                              const float* __restrict__ trel,
                                              const float* __restrict__ Wo,
                                              const float* __restrict__ bo,
                                              const float* __restrict__ Wl1,
                                              const float* __restrict__ bl1,
                                              const float* __restrict__ Wl2,
                                              const float* __restrict__ bl2,
                                              float* __restrict__ out) {
  const int b = blockIdx.x, j = threadIdx.x;
  __shared__ float conv[64];
  __shared__ float l1[16];
  const int s = srcn[b], t = tgtn[b];
  float as = bo[j], at = bo[j];
  for (int k = 0; k < 128; ++k) {
    float w = Wo[j * 128 + k];
    as += gout[(size_t)s * 128 + k] * w;
    at += gout[(size_t)t * 128 + k] * w;
  }
  as = fmaxf(as, 0.f);
  at = fmaxf(at, 0.f);
  conv[j] = tanhf(as + trel[b * 64 + j] - at);
  __syncthreads();
  if (j < 16) {
    float a = bl1[j];
    for (int k = 0; k < 64; ++k) a += conv[k] * Wl1[j * 64 + k];
    l1[j] = a;
  }
  __syncthreads();
  if (j == 0) {
    float o = bl2[0];
    for (int k = 0; k < 16; ++k) o += l1[k] * Wl2[k];
    out[b] = o;
  }
}

extern "C" void kernel_launch(void* const* d_in, const int* in_sizes, int n_in,
                              void* d_out, int out_size, void* d_ws, size_t ws_size,
                              hipStream_t stream) {
  (void)in_sizes; (void)n_in; (void)out_size; (void)ws_size;
  const float* node_feat   = (const float*)d_in[0];
  const int*   edge_type   = (const int*)d_in[1];
  const int*   graph_label = (const int*)d_in[2];
  const int*   edge_src    = (const int*)d_in[3];
  const int*   edge_dst    = (const int*)d_in[4];
  const int*   source_node = (const int*)d_in[5];
  const int*   target_node = (const int*)d_in[6];
  const float* rel_emb     = (const float*)d_in[7];
  const float* W_r2e       = (const float*)d_in[8];
  const float* b_r2e       = (const float*)d_in[9];
  const float* W_i_node    = (const float*)d_in[10];
  const float* W_i_edge    = (const float*)d_in[11];
  const float* W_att1      = (const float*)d_in[12];
  const float* W_att2      = (const float*)d_in[13];
  const float* W_h_node    = (const float*)d_in[14];
  const float* W_h_edge    = (const float*)d_in[15];
  const float* W_A1        = (const float*)d_in[16];
  const float* W_A2        = (const float*)d_in[17];
  const float* W_comm      = (const float*)d_in[18];
  const float* gru_bias    = (const float*)d_in[19];
  const float* gru_w_ih    = (const float*)d_in[20];
  const float* gru_w_hh    = (const float*)d_in[21];
  const float* gru_b_ih    = (const float*)d_in[22];
  const float* gru_b_hh    = (const float*)d_in[23];
  const float* W_o         = (const float*)d_in[24];
  const float* b_o         = (const float*)d_in[25];
  const float* W_l1        = (const float*)d_in[26];
  const float* b_l1        = (const float*)d_in[27];
  const float* W_l2        = (const float*)d_in[28];
  const float* b_l2        = (const float*)d_in[29];
  float* out = (float*)d_out;

  float* w = (float*)d_ws;
  float* input_node   = w; w += (size_t)N_ * 64;
  float* message_node = w; w += (size_t)N_ * 64;
  float* agg          = w; w += (size_t)N_ * 64;
  float* node_h       = w; w += (size_t)N_ * 64;
  float* target_rel   = w; w += B_ * 64;
  float* tgt_rel_h    = w; w += B_ * 64;
  float* gpart        = w; w += B_ * 64;
  float* h0           = w; w += B_ * 64;
  float* attg         = w; w += 2 * B_;
  float* att0         = w; w += E_;
  float* input_edge   = w; w += (size_t)E_ * 64;
  float* message_edge = w; w += (size_t)E_ * 64;
  // GI / GRU output alias the input_edge region (dead once the GRU phase starts)
  float* gi      = input_edge;
  float* gru_out = input_edge + 2 * (size_t)N_ * 192;

  const size_t aggBytes = (size_t)N_ * 64 * sizeof(float);

  k_input_node<<<N_ / 128, 256, 0, stream>>>(node_feat, W_i_node, input_node, message_node);
  k_graph_prep<<<B_, 64, 0, stream>>>(graph_label, rel_emb, W_r2e, b_r2e, target_rel, tgt_rel_h);
  k_gpart<<<B_, 64, 0, stream>>>(source_node, target_node, message_node, tgt_rel_h, W_att1, gpart);
  k_edge_init<<<E_ / 128, 256, 0, stream>>>(node_feat, rel_emb, edge_src, edge_dst, edge_type,
                                            W_i_edge, W_att1, W_att2, gpart, input_edge, att0);
  // ----- depth iteration d = 0 (att is per-edge att0; message_edge == input_edge)
  hipMemsetAsync(agg, 0, aggBytes, stream);
  k_scatter<<<(E_ * 64) / 256, 256, 0, stream>>>(input_edge, att0, nullptr, edge_dst, agg);
  k_node_update<<<N_ / 128, 256, 0, stream>>>(message_node, agg, W_h_node, message_node);
  k_edge_update<<<E_ / 128, 256, 0, stream>>>(edge_src, message_node, input_edge, W_h_edge, message_edge);
  k_graph_att<<<B_, 64, 0, stream>>>(source_node, target_node, message_node, tgt_rel_h,
                                     target_rel, W_A1, W_A2, attg);
  // ----- depth iteration d = 1 (att is per-graph attg[0..31])
  hipMemsetAsync(agg, 0, aggBytes, stream);
  k_scatter<<<(E_ * 64) / 256, 256, 0, stream>>>(message_edge, nullptr, attg, edge_dst, agg);
  k_node_update<<<N_ / 128, 256, 0, stream>>>(message_node, agg, W_h_node + 64 * 64, message_node);
  k_edge_update<<<E_ / 128, 256, 0, stream>>>(edge_src, message_node, input_edge,
                                              W_h_edge + 64 * 64, message_edge);
  k_graph_att<<<B_, 64, 0, stream>>>(source_node, target_node, message_node, tgt_rel_h,
                                     target_rel, W_A1 + 64 * 128, W_A2 + 64, attg + B_);
  // ----- final aggregation + communicate MLP
  hipMemsetAsync(agg, 0, aggBytes, stream);
  k_scatter<<<(E_ * 64) / 256, 256, 0, stream>>>(message_edge, nullptr, attg + B_, edge_dst, agg);
  k_comm<<<N_ / 128, 256, 0, stream>>>(agg, message_node, input_node, W_comm, node_h);
  // ----- GRU phase
  k_h0<<<B_, 256, 0, stream>>>(node_h, h0);
  k_gi<<<dim3(N_ / 128, 2), 256, 0, stream>>>(node_h, gru_bias, gru_w_ih, gru_b_ih, gi);
  k_gru_seq<<<dim3(B_, 2), 64, 0, stream>>>(gi, gru_w_hh, gru_b_hh, h0, gru_out);
  // ----- head
  k_final<<<B_, 64, 0, stream>>>(source_node, target_node, gru_out, target_rel,
                                 W_o, b_o, W_l1, b_l1, W_l2, b_l2, out);
}
